// Multi_Head_Self_Attention_35811437314560
// MI455X (gfx1250) — matrix operand order repaired and checked by reading, not compiled, not run on hardware
//
#include <hip/hip_runtime.h>
#include <hip/hip_bf16.h>
#include <math.h>

// ---------------------------------------------------------------------------
// Problem constants (from the reference)
// ---------------------------------------------------------------------------
#define NN   20000
#define HID  256
#define HEADS 8
#define DH   32
#define DEG  32
#define KTOP 16

typedef __bf16 bf16_t;
typedef __attribute__((ext_vector_type(16))) __bf16 v16bf;
typedef __attribute__((ext_vector_type(8)))  __bf16 v8bf;
typedef __attribute__((ext_vector_type(8)))  float  v8f;

// ---------------------------------------------------------------------------
// f32 -> bf16 conversion (one element per thread)
// ---------------------------------------------------------------------------
__global__ void cvt_f32_bf16(const float* __restrict__ src,
                             bf16_t* __restrict__ dst, int n) {
  int i = blockIdx.x * blockDim.x + threadIdx.x;
  if (i < n) dst[i] = (bf16_t)src[i];
}

// ---------------------------------------------------------------------------
// GEMM: C[M x 256] = A[M x 256] @ W^T + bias, optional leaky-relu epilogue.
//   A: bf16 row-major, W: bf16 row-major [out=256][in=256], bias/C: f32.
// Block = 256 threads = 8 waves. Block tile = 16 rows x 128 cols
// (each wave owns one 16x16 output tile). grid = (M/16, 256/128).
// K loop: 8 steps of v_wmma_f32_16x16x32_bf16.
// ---------------------------------------------------------------------------
__global__ __launch_bounds__(256)
void gemm_bf16_wmma(const bf16_t* __restrict__ A,
                    const bf16_t* __restrict__ W,
                    const float*  __restrict__ bias,
                    float*        __restrict__ C,
                    int leaky) {
  const int lane = threadIdx.x & 31;
  const int wave = threadIdx.x >> 5;            // 0..7
  const int row0 = blockIdx.x * 16;
  const int col0 = blockIdx.y * 128 + wave * 16;

  const int r  = lane & 15;                     // row (A) / col (B,C) within tile
  const int hi = lane >> 4;                     // 0 or 1 (K-half select)

  // A-fragment source: lane r holds row row0+r; elements 0..15 = K hi*16+0..15
  // (contiguous K order, matching B).
  const bf16_t* aRow = A + (size_t)(row0 + r) * HID + hi * 16;
  // B-fragment source: lane r holds output column col0+r; B[k][n] = W[n][k],
  // so K walks 16 contiguous halfwords of W's row (contiguous K order, as in A:
  // lanes 0-15 hold K=0..15, lanes 16-31 hold K=16..31).
  const bf16_t* bRow = W + (size_t)(col0 + r) * HID + hi * 16;

  v8f acc = {};
  #pragma unroll
  for (int kb = 0; kb < 8; ++kb) {
    // prefetch the next K tile of A (global_prefetch_b8)
    if (kb < 7) __builtin_prefetch(aRow + (kb + 1) * 32, 0, 1);

    v8bf alo = *(const v8bf*)(aRow + kb * 32);
    v8bf ahi = *(const v8bf*)(aRow + kb * 32 + 8);
    v16bf a, b;
    #pragma unroll
    for (int e = 0; e < 8; ++e) { a[e] = alo[e]; a[e + 8] = ahi[e]; }
    b = *(const v16bf*)(bRow + kb * 32);

    acc = __builtin_amdgcn_wmma_f32_16x16x32_bf16(
        /*neg_a=*/false, a, /*neg_b=*/false, b,
        /*c_mod=*/(short)0, acc, /*reuse_a=*/false, /*reuse_b=*/false);
  }

  // Epilogue. C/D layout: VGPR e -> row (e + 8*hi), col = lane&15.
  const float bv = bias[col0 + r];
  #pragma unroll
  for (int e = 0; e < 8; ++e) {
    const int m = e + 8 * hi;
    float o = acc[e] + bv;
    if (leaky) o = (o > 0.f) ? o : 0.01f * o;
    C[(size_t)(row0 + m) * HID + (col0 + r)] = o;
  }
}

// ---------------------------------------------------------------------------
// Per-node neighbor attention. One block per node; wave w = head w.
// Phase 1 (lane = neighbor): top-16 weight mask + normalize, score dot,
//   wave-wide softmax over the 32 neighbors via shfl reductions.
// Phase 2 (lane = channel): agg[c] = sum_d attn[d] * v[nbr[d]][c].
// k/v rows (20 MB each) live in the 192 MB L2, so gathers are L2 hits.
// ---------------------------------------------------------------------------
__global__ __launch_bounds__(256)
void graph_attn(const float* __restrict__ q,
                const float* __restrict__ k,
                const float* __restrict__ v,
                const int*   __restrict__ nbr,
                const float* __restrict__ ew,
                bf16_t*      __restrict__ agg) {
  __shared__ float s_q[HID];
  __shared__ float s_w[DEG];
  __shared__ int   s_n[DEG];

  const int n = blockIdx.x;
  const int t = threadIdx.x;
  s_q[t] = q[(size_t)n * HID + t];
  if (t < DEG) {
    s_w[t] = ew[(size_t)n * DEG + t];
    s_n[t] = nbr[(size_t)n * DEG + t];
  }
  __syncthreads();

  const int lane = t & 31;
  const int head = t >> 5;

  // ---- top-k rank (stable argsort tie-break: greater, or equal & lower idx)
  const float wd = s_w[lane];
  int rank = 0;
  #pragma unroll
  for (int j = 0; j < DEG; ++j) {
    const float wj = s_w[j];
    rank += (wj > wd) || (wj == wd && j < lane);
  }
  float wm = (rank < KTOP) ? wd : 0.f;
  float wsum = wm;
  #pragma unroll
  for (int off = 16; off; off >>= 1) wsum += __shfl_xor(wsum, off, 32);
  const float wn = wm / (wsum + 1e-5f);

  // ---- score: <k[nbr[lane]], q[n]> over this head's 32 channels
  const float* krow = k + (size_t)s_n[lane] * HID + head * DH;
  const float* qrow = &s_q[head * DH];
  float sc = 0.f;
  #pragma unroll
  for (int c = 0; c < DH; c += 4) {
    const float4 kv = *(const float4*)(krow + c);
    const float4 qv = *(const float4*)(qrow + c);
    sc += kv.x * qv.x + kv.y * qv.y + kv.z * qv.z + kv.w * qv.w;
  }
  const float logit = sc * wn * 0.17677669529663687f;  // 1/sqrt(32)

  // ---- softmax across the 32 lanes (= 32 neighbors)
  float mx = logit;
  #pragma unroll
  for (int off = 16; off; off >>= 1) mx = fmaxf(mx, __shfl_xor(mx, off, 32));
  const float ex = __expf(logit - mx);
  float es = ex;
  #pragma unroll
  for (int off = 16; off; off >>= 1) es += __shfl_xor(es, off, 32);
  const float attn = ex / es;

  // ---- aggregation: lane now owns channel head*32+lane
  float accv = 0.f;
  for (int d = 0; d < DEG; ++d) {
    const float a = __shfl(attn, d, 32);
    accv += a * v[(size_t)s_n[d] * HID + head * DH + lane];
  }
  agg[(size_t)n * HID + head * DH + lane] = (bf16_t)accv;
}

// ---------------------------------------------------------------------------
// Host-side orchestration
// ---------------------------------------------------------------------------
extern "C" void kernel_launch(void* const* d_in, const int* in_sizes, int n_in,
                              void* d_out, int out_size, void* d_ws, size_t ws_size,
                              hipStream_t stream) {
  const float* h   = (const float*)d_in[0];
  const int*   nbr = (const int*)  d_in[1];
  const float* ew  = (const float*)d_in[2];
  const float* Wq  = (const float*)d_in[3];
  const float* bq  = (const float*)d_in[4];
  const float* Wk  = (const float*)d_in[5];
  const float* bk  = (const float*)d_in[6];
  const float* Wv  = (const float*)d_in[7];
  const float* bv  = (const float*)d_in[8];
  const float* Wo  = (const float*)d_in[9];
  const float* bo  = (const float*)d_in[10];
  float* out = (float*)d_out;

  // workspace carve-out (all 256B aligned)
  size_t off = 0;
  char* ws = (char*)d_ws;
  auto carve = [&](size_t bytes) -> void* {
    void* p = ws + off;
    off = (off + bytes + 255) & ~(size_t)255;
    return p;
  };
  float*  qf    = (float*) carve((size_t)NN * HID * sizeof(float));
  float*  kf    = (float*) carve((size_t)NN * HID * sizeof(float));
  float*  vf    = (float*) carve((size_t)NN * HID * sizeof(float));
  bf16_t* hbf   = (bf16_t*)carve((size_t)NN * HID * sizeof(bf16_t));
  bf16_t* aggbf = (bf16_t*)carve((size_t)NN * HID * sizeof(bf16_t));
  bf16_t* Wqbf  = (bf16_t*)carve((size_t)HID * HID * sizeof(bf16_t));
  bf16_t* Wkbf  = (bf16_t*)carve((size_t)HID * HID * sizeof(bf16_t));
  bf16_t* Wvbf  = (bf16_t*)carve((size_t)HID * HID * sizeof(bf16_t));
  bf16_t* Wobf  = (bf16_t*)carve((size_t)HID * HID * sizeof(bf16_t));
  (void)ws_size; (void)n_in; (void)in_sizes; (void)out_size;

  // 1) bf16 conversions
  {
    const int nh = NN * HID, nw = HID * HID;
    cvt_f32_bf16<<<(nh + 255) / 256, 256, 0, stream>>>(h,  hbf,  nh);
    cvt_f32_bf16<<<(nw + 255) / 256, 256, 0, stream>>>(Wq, Wqbf, nw);
    cvt_f32_bf16<<<(nw + 255) / 256, 256, 0, stream>>>(Wk, Wkbf, nw);
    cvt_f32_bf16<<<(nw + 255) / 256, 256, 0, stream>>>(Wv, Wvbf, nw);
    cvt_f32_bf16<<<(nw + 255) / 256, 256, 0, stream>>>(Wo, Wobf, nw);
  }

  // 2) q/k/v projections: WMMA bf16 GEMM, f32 output
  dim3 ggrid(NN / 16, HID / 128);  // 1250 x 2
  gemm_bf16_wmma<<<ggrid, 256, 0, stream>>>(hbf, Wqbf, bq, qf, 0);
  gemm_bf16_wmma<<<ggrid, 256, 0, stream>>>(hbf, Wkbf, bk, kf, 0);
  gemm_bf16_wmma<<<ggrid, 256, 0, stream>>>(hbf, Wvbf, bv, vf, 0);

  // 3) per-node neighbor attention -> agg (bf16 for the O projection)
  graph_attn<<<NN, 256, 0, stream>>>(qf, kf, vf, nbr, ew, aggbf);

  // 4) output projection + leaky-relu epilogue -> d_out (f32)
  gemm_bf16_wmma<<<ggrid, 256, 0, stream>>>(aggbf, Wobf, bo, out, 1);
}
